// MultiHeadAttention_90735479095531
// MI455X (gfx1250) — compile-verified, matrix operand order checked by reading, not run on hardware
//
#include <hip/hip_runtime.h>
#include <hip/hip_bf16.h>

// ---------------------------------------------------------------------------
// MHA forward for MI455X (gfx1250): bf16 WMMA (f32 accum) everywhere,
// Tensor Data Mover (TDM) staging for the pure-copy tiles.
//   B=4, QN=N=2048, D=1024, H=16, dk=64.
//   ws layout (bf16): Qp | Kp | Vp  [B][H][seq][64]  +  ctx [B][QN][1024]
// ---------------------------------------------------------------------------

typedef __attribute__((ext_vector_type(16))) __bf16 v16bf;
typedef __attribute__((ext_vector_type(8)))  float  v8f;

#if defined(__has_builtin)
#  if __has_builtin(__builtin_amdgcn_tensor_load_to_lds) && \
      __has_builtin(__builtin_amdgcn_s_wait_tensorcnt)
#    define HAVE_TDM 1
#  endif
#endif
#ifndef HAVE_TDM
#  define HAVE_TDM 0
#endif

union Frag {            // 8 VGPRs: one 16x32 bf16 WMMA operand
    v16bf    bf;
    uint32_t u[8];
    uint4    q[2];
};

__device__ __forceinline__ uint32_t f32_to_bf16_bits(float f) {
    uint32_t u = __builtin_bit_cast(uint32_t, f);
    u += 0x7FFFu + ((u >> 16) & 1u);     // round-to-nearest-even
    return u >> 16;
}
__device__ __forceinline__ uint32_t pack2(float a, float b) {
    return f32_to_bf16_bits(a) | (f32_to_bf16_bits(b) << 16);
}

__device__ __forceinline__ v8f wmma_bf16(const Frag& a, const Frag& b, v8f c) {
    // v_wmma_f32_16x16x32_bf16  (neg_a, A, neg_b, B, c_mod, C, reuse_a, reuse_b)
    return __builtin_amdgcn_wmma_f32_16x16x32_bf16(false, a.bf, false, b.bf,
                                                   (short)0, c, false, false);
}

// Load one 16x32 bf16 operand from a row-major [row][rowStrideU*2 bf16] array.
// ISA 16-bit A layout: lanes 0-15 hold K=0..7,16..23; lanes 16-31 hold K=8..15,24..31.
__device__ __forceinline__ void load_frag_u(Frag& f, const uint32_t* base,
                                            int row, int rowStrideU, int fragK, int lane) {
    const int r = lane & 15;
    const int h = (lane >> 4) & 1;
    const uint32_t* p = base + (size_t)(row + r) * rowStrideU + fragK * 16 + h * 4;
    f.q[0] = *(const uint4*)(p);
    f.q[1] = *(const uint4*)(p + 8);
}

#if HAVE_TDM
// ---------------------------------------------------------------------------
// Tensor Data Mover: D# group0/group1 per CDNA5 ISA.
//  group0: [1:0]=count=1, [63:32]=lds_addr, [120:64]=global_addr, [127:126]=type=2
//  group1: d0[17:16]=data_size(2 -> 4B); tensor_dim0 bits 79:48; tensor_dim1 111:80;
//          tile_dim0 127:112; tile_dim1 143:128; tensor_dim0_stride 207:160.
// Copies a 2D tile of wDw dwords x hRows rows (row stride strideDw dwords)
// from global to LDS, densely packed.  hRows==1 degenerates to a 1D copy.
// ---------------------------------------------------------------------------
typedef __attribute__((ext_vector_type(4))) unsigned int u32x4;
typedef __attribute__((ext_vector_type(4))) int          i32x4;
typedef __attribute__((ext_vector_type(8))) int          i32x8;

__device__ __forceinline__ void tdm_load_2d(uint32_t lds_addr, const void* gp,
                                            uint32_t wDw, uint32_t hRows,
                                            uint32_t strideDw) {
    const uint64_t ga = (uint64_t)(uintptr_t)gp;
    u32x4 g0;
    g0[0] = 1u;                                              // count=1 (user mode)
    g0[1] = lds_addr;                                        // LDS byte address
    g0[2] = (uint32_t)ga;                                    // global_addr[31:0]
    g0[3] = (uint32_t)((ga >> 32) & 0x1FFFFFFu) | (2u << 30);// ga[56:32] | type=2
    i32x8 g1 = (i32x8)0;
    g1[0] = (int)(2u << 16);                                 // data_size = 4 bytes
    g1[1] = (int)((wDw & 0xFFFFu) << 16);                    // tensor_dim0[15:0]
    g1[2] = (int)((wDw >> 16) | ((hRows & 0xFFFFu) << 16));  // td0[31:16] | td1[15:0]
    g1[3] = (int)((wDw & 0xFFFFu) << 16);                    // td1[31:16]=0 | tile_dim0
    g1[4] = (int)(hRows & 0xFFFFu);                          // tile_dim1 | tile_dim2=0
    g1[5] = (int)strideDw;                                   // tensor_dim0_stride[31:0]
    const i32x4 z4 = (i32x4)0;
#if defined(__clang_major__) && __clang_major__ >= 23
    __builtin_amdgcn_tensor_load_to_lds(g0, g1, z4, z4, (i32x8)0, 0);
#else
    __builtin_amdgcn_tensor_load_to_lds(g0, g1, z4, z4, 0);
#endif
}
__device__ __forceinline__ void tdm_wait0() {
    __builtin_amdgcn_s_wait_tensorcnt(0);
}
__device__ __forceinline__ uint32_t lds_addr_of(const void* p) {
    return (uint32_t)(uintptr_t)p;      // low 32 bits of generic addr = LDS offset
}
#endif  // HAVE_TDM

// ---------------------------------------------------------------------------
// GEMM: out[m][e] = sum_d A[m][d] * W[e][d]    (y = x @ W.T, bias-free Linear)
// A_IS_BF16: A is bf16 (ctx, TDM-staged) else f32 (packed to bf16 in LDS).
// OUT_HEADS: write bf16 head-major, else write f32 row-major.
// Workgroup tile 128x128, BK=32, 8 waves (2 m x 4 n), each wave 64x32.
// ---------------------------------------------------------------------------
template <bool A_IS_BF16, bool OUT_HEADS>
__global__ __launch_bounds__(256) void gemm_xwT(const void* __restrict__ Aptr,
                                                const float* __restrict__ W,
                                                void* __restrict__ out,
                                                int M, int K, int Nn,
                                                int seq, int heads) {
    const int tid  = threadIdx.x;
    const int lane = tid & 31;
    const int wid  = tid >> 5;
    const int wm   = wid >> 2;           // 0..1
    const int wn   = wid & 3;            // 0..3
    const int m0   = blockIdx.y * 128;
    const int n0   = blockIdx.x * 128;

    __shared__ uint32_t lA[128 * 16];    // 128 x 32 bf16
    __shared__ uint32_t lB[128 * 16];

    v8f acc[4][2] = {};

    const int row  = tid >> 1;           // 0..127 staging row
    const int half = tid & 1;            // k-half 0..15 / 16..31

    for (int kt = 0; kt < K; kt += 32) {
        // ---- stage A tile ----
        if (A_IS_BF16) {
#if HAVE_TDM
            if (wid == 0) {              // one TDM descriptor per k-step, wave 0 only
                const uint32_t* Ag0 = (const uint32_t*)Aptr +
                                      (size_t)m0 * (K >> 1) + (kt >> 1);
                tdm_load_2d(lds_addr_of(lA), Ag0, 16u, 128u, (uint32_t)(K >> 1));
                tdm_wait0();
            }
#else
            const uint32_t* Ag = (const uint32_t*)Aptr +
                                 (size_t)(m0 + row) * (K >> 1) + (kt >> 1) + half * 8;
            *(uint4*)&lA[row * 16 + half * 8]     = *(const uint4*)(Ag);
            *(uint4*)&lA[row * 16 + half * 8 + 4] = *(const uint4*)(Ag + 4);
#endif
        } else {
            const float* Ag = (const float*)Aptr + (size_t)(m0 + row) * K + kt + half * 16;
            float4 f0 = *(const float4*)(Ag);
            float4 f1 = *(const float4*)(Ag + 4);
            float4 f2 = *(const float4*)(Ag + 8);
            float4 f3 = *(const float4*)(Ag + 12);
            uint32_t* d = &lA[row * 16 + half * 8];
            d[0] = pack2(f0.x, f0.y); d[1] = pack2(f0.z, f0.w);
            d[2] = pack2(f1.x, f1.y); d[3] = pack2(f1.z, f1.w);
            d[4] = pack2(f2.x, f2.y); d[5] = pack2(f2.z, f2.w);
            d[6] = pack2(f3.x, f3.y); d[7] = pack2(f3.z, f3.w);
        }
        // ---- stage B tile (W rows e = n0+row), f32 -> bf16 pack ----
        {
            const float* Bg = W + (size_t)(n0 + row) * K + kt + half * 16;
            float4 f0 = *(const float4*)(Bg);
            float4 f1 = *(const float4*)(Bg + 4);
            float4 f2 = *(const float4*)(Bg + 8);
            float4 f3 = *(const float4*)(Bg + 12);
            uint32_t* d = &lB[row * 16 + half * 8];
            d[0] = pack2(f0.x, f0.y); d[1] = pack2(f0.z, f0.w);
            d[2] = pack2(f1.x, f1.y); d[3] = pack2(f1.z, f1.w);
            d[4] = pack2(f2.x, f2.y); d[5] = pack2(f2.z, f2.w);
            d[6] = pack2(f3.x, f3.y); d[7] = pack2(f3.z, f3.w);
            if (kt + 32 < K) {   // global_prefetch_b8 for next k-slab
                __builtin_prefetch((const void*)(Bg + 32), 0, 1);
                if (!A_IS_BF16)
                    __builtin_prefetch((const void*)((const float*)Aptr +
                                       (size_t)(m0 + row) * K + kt + 32 + half * 16), 0, 1);
            }
        }
        __syncthreads();

        Frag af[4], bfr[2];
#pragma unroll
        for (int i = 0; i < 4; ++i) load_frag_u(af[i],  lA, wm * 64 + i * 16, 16, 0, lane);
#pragma unroll
        for (int j = 0; j < 2; ++j) load_frag_u(bfr[j], lB, wn * 32 + j * 16, 16, 0, lane);
#pragma unroll
        for (int i = 0; i < 4; ++i)
#pragma unroll
            for (int j = 0; j < 2; ++j)
                acc[i][j] = wmma_bf16(af[i], bfr[j], acc[i][j]);
        __syncthreads();
    }

    // ---- epilogue: C/D layout VGPR r -> row r (lanes 0-15) / r+8 (lanes 16-31) ----
    const int rh = (lane >> 4) & 1;
#pragma unroll
    for (int i = 0; i < 4; ++i)
#pragma unroll
        for (int j = 0; j < 2; ++j)
#pragma unroll
            for (int r = 0; r < 8; ++r) {
                const int mrow = m0 + wm * 64 + i * 16 + r + rh * 8;
                const int ncol = n0 + wn * 32 + j * 16 + (lane & 15);
                const float val = acc[i][j][r];
                if (OUT_HEADS) {
                    const int b = mrow / seq, s = mrow % seq;
                    const int h = ncol >> 6, c = ncol & 63;
                    ((uint16_t*)out)[((((size_t)b * heads + h) * seq + s) << 6) + c] =
                        (uint16_t)f32_to_bf16_bits(val);
                } else {
                    ((float*)out)[(size_t)mrow * Nn + ncol] = val;
                }
            }
}

// ---------------------------------------------------------------------------
// Flash attention: grid (QN/128, B*H), 256 threads (8 waves).
// Each wave owns 16 q-rows; K tile 128x64 + V^T tile 64x128 staged in LDS.
// With TDM: K and raw-V tiles are DMA'd by one descriptor each (wave 0),
// V transpose becomes an LDS->LDS pass.
// ---------------------------------------------------------------------------
__global__ __launch_bounds__(256) void flash_attn(const uint32_t* __restrict__ Qp,
                                                  const uint32_t* __restrict__ Kp,
                                                  const uint16_t* __restrict__ Vp,
                                                  const int* __restrict__ mask,
                                                  uint16_t* __restrict__ ctx,
                                                  int B, int H, int QN, int N) {
    const int tid  = threadIdx.x;
    const int lane = tid & 31;
    const int w    = tid >> 5;
    const int bh   = blockIdx.y;
    const int b    = bh / H, h = bh % H;
    const int q0   = blockIdx.x * 128;
    const int qrowBase = q0 + w * 16;

    __shared__ uint32_t lK[128 * 32];        // 128 x 64 bf16  (16 KB)
#if HAVE_TDM
    __shared__ uint32_t lVraw[128 * 32];     // raw V tile (TDM dest, 16 KB)
#endif
    __shared__ uint32_t lVt[64 * 64];        // V^T: 64 x 128 bf16 (16 KB)
    __shared__ uint32_t lP[8][16 * 64];      // per-wave P: 16 x 128 bf16 (32 KB)

    // Q fragments (row stride = 32 uints = 64 bf16), kept in VGPRs all kernel.
    Frag qf[2];
    const uint32_t* Qbase = Qp + (size_t)bh * QN * 32;
#pragma unroll
    for (int f = 0; f < 2; ++f) load_frag_u(qf[f], Qbase, qrowBase, 32, f, lane);

    float mrow[8], lrow[8];
    v8f   oacc[4] = {};
#pragma unroll
    for (int r = 0; r < 8; ++r) { mrow[r] = -3.0e38f; lrow[r] = 0.f; }

    const size_t kvBase = (size_t)bh * N * 32;   // uints
    const int colBase = lane & 15;
    const int rh = (lane >> 4) & 1;

    for (int n0 = 0; n0 < N; n0 += 128) {
#if HAVE_TDM
        // ---- TDM: DMA K tile and raw V tile into LDS (wave 0 issues) ----
        if (w == 0) {
            tdm_load_2d(lds_addr_of(lK),   Kp + kvBase + (size_t)n0 * 32, 4096u, 1u, 4096u);
            tdm_load_2d(lds_addr_of(lVraw),
                        (const uint32_t*)(Vp + ((size_t)bh * N + n0) * 64), 4096u, 1u, 4096u);
            tdm_wait0();                         // s_wait_tensorcnt 0
        }
        __syncthreads();
        // ---- transpose V from LDS: lVt[c][n] ----
#pragma unroll
        for (int it = 0; it < 4; ++it) {
            const int idx = tid + it * 256;
            const int n = idx >> 3, c0 = (idx & 7) * 8;
            uint4 d = *(const uint4*)&lVraw[(size_t)n * 32 + (idx & 7) * 4];
            const uint16_t* e = (const uint16_t*)&d;
#pragma unroll
            for (int c = 0; c < 8; ++c)
                ((uint16_t*)lVt)[(c0 + c) * 128 + n] = e[c];
        }
#else
        // ---- stage K tile (flat 1024 uint4 copy) ----
        {
            const uint4* src = (const uint4*)(Kp + kvBase + (size_t)n0 * 32);
            uint4* dst = (uint4*)lK;
#pragma unroll
            for (int it = 0; it < 4; ++it) dst[tid + it * 256] = src[tid + it * 256];
        }
        // ---- stage V tile transposed from global: lVt[c][n] ----
        {
            const uint16_t* vsrc = Vp + ((size_t)bh * N + n0) * 64;
#pragma unroll
            for (int it = 0; it < 4; ++it) {
                const int idx = tid + it * 256;
                const int n = idx >> 3, c0 = (idx & 7) * 8;
                uint4 d = *(const uint4*)(vsrc + (size_t)n * 64 + c0);
                const uint16_t* e = (const uint16_t*)&d;
#pragma unroll
                for (int c = 0; c < 8; ++c)
                    ((uint16_t*)lVt)[(c0 + c) * 128 + n] = e[c];
            }
        }
#endif
        if (n0 + 128 < N)
            __builtin_prefetch((const void*)(Kp + kvBase + (size_t)(n0 + 128) * 32 + tid * 4), 0, 1);
        __syncthreads();

        // ---- S = Q @ K^T (16 x 128 per wave) ----
        v8f sacc[8];
#pragma unroll
        for (int j = 0; j < 8; ++j) {
            Frag kb;
            v8f a = {};
            load_frag_u(kb, lK, j * 16, 32, 0, lane);
            a = wmma_bf16(qf[0], kb, a);
            load_frag_u(kb, lK, j * 16, 32, 1, lane);
            a = wmma_bf16(qf[1], kb, a);
            sacc[j] = a;
        }
        // ---- scale + mask ----
#pragma unroll
        for (int j = 0; j < 8; ++j)
#pragma unroll
            for (int r = 0; r < 8; ++r) {
                const int q = qrowBase + r + rh * 8;
                const int n = n0 + j * 16 + colBase;
                float s = sacc[j][r] * 0.125f;                 // 1/sqrt(64)
                if (mask[((size_t)b * QN + q) * N + n] == 0) s = -1.0e9f;
                sacc[j][r] = s;
            }
        // ---- running softmax: row max via 16-lane xor-shuffles ----
        float mnew[8], alpha[8], lsum[8];
#pragma unroll
        for (int r = 0; r < 8; ++r) {
            float mx = sacc[0][r];
#pragma unroll
            for (int j = 1; j < 8; ++j) mx = fmaxf(mx, sacc[j][r]);
#pragma unroll
            for (int s = 1; s < 16; s <<= 1) mx = fmaxf(mx, __shfl_xor(mx, s, 32));
            mnew[r]  = fmaxf(mrow[r], mx);
            alpha[r] = __expf(mrow[r] - mnew[r]);
            mrow[r]  = mnew[r];
            lsum[r]  = 0.f;
        }
        // ---- P = exp(S - m); write per-wave LDS slab (A-layout-friendly) ----
#pragma unroll
        for (int j = 0; j < 8; ++j)
#pragma unroll
            for (int r = 0; r < 8; ++r) {
                const float p = __expf(sacc[j][r] - mnew[r]);
                lsum[r] += p;
                ((uint16_t*)lP[w])[(r + rh * 8) * 128 + j * 16 + colBase] =
                    (uint16_t)f32_to_bf16_bits(p);
            }
#pragma unroll
        for (int r = 0; r < 8; ++r) {
            float ls = lsum[r];
#pragma unroll
            for (int s = 1; s < 16; s <<= 1) ls += __shfl_xor(ls, s, 32);
            lrow[r] = lrow[r] * alpha[r] + ls;
#pragma unroll
            for (int jj = 0; jj < 4; ++jj) oacc[jj][r] *= alpha[r];
        }
        // ---- O += P @ V  (P from LDS as A-frags, V^T as B-frags) ----
        Frag pf[4];
#pragma unroll
        for (int kk = 0; kk < 4; ++kk) load_frag_u(pf[kk], lP[w], 0, 64, kk, lane);
#pragma unroll
        for (int jj = 0; jj < 4; ++jj) {
            v8f a = oacc[jj];
#pragma unroll
            for (int kk = 0; kk < 4; ++kk) {
                Frag vf;
                load_frag_u(vf, lVt, jj * 16, 64, kk, lane);
                a = wmma_bf16(pf[kk], vf, a);
            }
            oacc[jj] = a;
        }
        __syncthreads();
    }

    // ---- epilogue: ctx[b][q][h*64+c] bf16 ----
#pragma unroll
    for (int jj = 0; jj < 4; ++jj)
#pragma unroll
        for (int r = 0; r < 8; ++r) {
            const float v = oacc[jj][r] / lrow[r];
            const int q = qrowBase + r + rh * 8;
            const int c = jj * 16 + (lane & 15);
            ctx[((size_t)b * QN + q) * 1024 + h * 64 + c] = (uint16_t)f32_to_bf16_bits(v);
        }
}

// ---------------------------------------------------------------------------
extern "C" void kernel_launch(void* const* d_in, const int* in_sizes, int n_in,
                              void* d_out, int out_size, void* d_ws, size_t ws_size,
                              hipStream_t stream) {
    (void)in_sizes; (void)n_in; (void)out_size; (void)ws_size;
    const float* q    = (const float*)d_in[0];
    const float* k    = (const float*)d_in[1];
    const float* v    = (const float*)d_in[2];
    const int*   mask = (const int*)d_in[3];
    const float* Wq   = (const float*)d_in[4];
    const float* Wk   = (const float*)d_in[5];
    const float* Wv   = (const float*)d_in[6];
    const float* Wo   = (const float*)d_in[7];

    const int Bsz = 4, H = 16, QN = 2048, N = 2048, D = 1024, DK = 64;
    const size_t headElems = (size_t)Bsz * H * QN * DK;      // 8,388,608 bf16 each

    uint16_t* Qp  = (uint16_t*)d_ws;
    uint16_t* Kp  = Qp + headElems;
    uint16_t* Vp  = Kp + headElems;
    uint16_t* ctx = Vp + headElems;                          // B*QN*D bf16

    dim3 gGemm(D / 128, (Bsz * QN) / 128);                   // (8, 64)
    gemm_xwT<false, true><<<gGemm, 256, 0, stream>>>((const void*)q, Wq, (void*)Qp,
                                                     Bsz * QN, D, D, QN, H);
    gemm_xwT<false, true><<<gGemm, 256, 0, stream>>>((const void*)k, Wk, (void*)Kp,
                                                     Bsz * QN, D, D, QN, H);
    gemm_xwT<false, true><<<gGemm, 256, 0, stream>>>((const void*)v, Wv, (void*)Vp,
                                                     Bsz * QN, D, D, QN, H);

    dim3 gAttn(QN / 128, Bsz * H);                           // (16, 64)
    flash_attn<<<gAttn, 256, 0, stream>>>((const uint32_t*)Qp, (const uint32_t*)Kp,
                                          (const uint16_t*)Vp, mask, ctx,
                                          Bsz, H, QN, N);

    gemm_xwT<true, false><<<gGemm, 256, 0, stream>>>((const void*)ctx, Wo, d_out,
                                                     Bsz * QN, D, D, QN, H);
}
